// LatentRolloutHead_5789615915731
// MI455X (gfx1250) — compile-verified, matrix operand order checked by reading, not run on hardware
//
#include <hip/hip_runtime.h>
#include <math.h>

// ---------------------------------------------------------------------------
// LatentRolloutHead — GRU rollout + GELU MLP head, bf16 WMMA, gfx1250 (CDNA5)
// B=16, T=128, D=H=1024, horizon=16.
// - gate-interleaved fused GRU-cell GEMM (register-only epilogue)
// - async global->LDS double-buffered pipeline, unroll-by-2, always-prefetch
// ---------------------------------------------------------------------------

#define BATCH   16
#define TSEQ    128
#define DDIM    1024
#define HDIM    1024
#define RH      16
#define SC      112            // start_count = T - rh
#define MROWS   (BATCH * SC)   // 1792
#define GK      1024           // shared K of every GEMM in this model

typedef __attribute__((ext_vector_type(16))) __bf16 bf16x16;
typedef __attribute__((ext_vector_type(8)))  float  f32x8;

__device__ __forceinline__ unsigned short f2bf(float f) {
    union { float f; unsigned u; } v; v.f = f;
    unsigned r = v.u + 0x7FFFu + ((v.u >> 16) & 1u);
    return (unsigned short)(r >> 16);
}

__device__ __forceinline__ float sigmoidf_(float x) {
    return 1.0f / (1.0f + expf(-x));
}

// ---- gfx1250 async global->LDS copy (ASYNCcnt-tracked, no VGPR staging) ----
__device__ __forceinline__ void async_b128(const void* gaddr, const void* lds) {
    asm volatile("global_load_async_to_lds_b128 %0, %1, off"
                 :: "v"((unsigned)(size_t)lds), "v"(gaddr) : "memory");
}
__device__ __forceinline__ void wait_async0() {
    asm volatile("s_wait_asynccnt 0" ::: "memory");
}

// ------------------------- conversion / init kernels ------------------------

__global__ void k_f32_to_bf16(const float* __restrict__ src,
                              unsigned short* __restrict__ dst, int n) {
    int i = blockIdx.x * blockDim.x + threadIdx.x;
    if (i < n) dst[i] = f2bf(src[i]);
}

// Fused gate weight, gate-interleaved row order:
// fused row c = g16*96 + gate*16 + u  (j = g16*16 + u), K = 1024
// gate 0..2 -> W_ih[:, :1024] rows gate*1024+j ; gate 3..5 -> W_hh rows (gate-3)*1024+j
__global__ void k_build_wg(const float* __restrict__ Wih,
                           const float* __restrict__ Whh,
                           unsigned short* __restrict__ Wg) {
    int i = blockIdx.x * blockDim.x + threadIdx.x;   // over 6144*1024
    if (i >= 6 * HDIM * DDIM) return;
    int c = i >> 10;
    int k = i & 1023;
    int g16 = c / 96;
    int rem = c - g16 * 96;
    int gg  = rem >> 4;
    int u   = rem & 15;
    int j   = g16 * 16 + u;
    float v = (gg < 3) ? Wih[(size_t)(gg * 1024 + j) * 2048 + k]
                       : Whh[(size_t)((gg - 3) * 1024 + j) * 1024 + k];
    Wg[(size_t)c * 1024 + k] = f2bf(v);
}

// Context half of W_ih (natural gate*1024+j order -> matches ctxproj layout)
__global__ void k_build_wc(const float* __restrict__ Wih,
                           unsigned short* __restrict__ Wc) {
    int i = blockIdx.x * blockDim.x + threadIdx.x;   // over 3072*1024
    if (i >= 3 * HDIM * DDIM) return;
    int n = i >> 10;
    int k = i & 1023;
    Wc[(size_t)n * 1024 + k] = f2bf(Wih[(size_t)n * 2048 + 1024 + k]);
}

// hidden0 = state_sequence[:, :112, :] -> f32 + bf16 copies
__global__ void k_init_h(const float* __restrict__ state,
                         float* __restrict__ hF,
                         unsigned short* __restrict__ hB) {
    int i = blockIdx.x * blockDim.x + threadIdx.x;   // over 1792*1024
    if (i >= MROWS * DDIM) return;
    int m = i >> 10;
    int j = i & 1023;
    int b = m / SC;
    int t = m - b * SC;
    float v = state[((size_t)b * TSEQ + t) * DDIM + j];
    hF[i] = v;
    hB[i] = f2bf(v);
}

// ------------------------------ generic GEMM -------------------------------
// C[M,N] = A[M,1024](bf16 rm) * Wrow[N,1024]^T (bf16 rm).  128x128x32 tiles,
// 8 waves (2x4), 64x32 wave tile, async-LDS unroll-by-2 double buffer.
// EPI 0: raw f32 (ldOut) | EPI 1: +bias, exact GELU -> bf16 | EPI 2: +bias,
// f32 store to d_out at ((m*RH+sIdx)*1024+col).

__device__ __forceinline__ bf16x16 load_frag(const unsigned short* smem,
                                             int row, int kb, int lda) {
    union { uint4 u[2]; bf16x16 v; } f;
    const unsigned short* p = smem + row * lda + kb;
    f.u[0] = *(const uint4*)(p);
    f.u[1] = *(const uint4*)(p + 16);
    return f.v;
}

template <int EPI>
__global__ __launch_bounds__(256) void k_gemm(
    const unsigned short* __restrict__ A,
    const unsigned short* __restrict__ Bw,
    float* __restrict__ outF,
    unsigned short* __restrict__ outB,
    const float* __restrict__ bias,
    int M, int N, int ldOut, int sIdx)
{
    constexpr int BM = 128, BN = 128, BK = 32, LDA = BK + 8, K = GK;
    constexpr unsigned GSTR = 64u * K * 2u;           // 64-row chunk stride (bytes)
    constexpr unsigned LSTR = 64u * LDA * 2u;         // matching LDS stride
    __shared__ __align__(16) unsigned short As[2][BM * LDA];
    __shared__ __align__(16) unsigned short Bs[2][BN * LDA];

    const int t    = threadIdx.x;
    const int m0   = blockIdx.y * BM;
    const int n0   = blockIdx.x * BN;
    const int lane = t & 31;
    const int wave = t >> 5;
    const int wm   = (wave >> 2) * 64;
    const int wn   = (wave & 3) * 32;
    const int lrow = lane & 15;
    const int kb   = (lane < 16) ? 0 : 8;

    const int rowL = t >> 2;
    const int colL = (t & 3) * 8;

    // per-thread base addresses (chunk 0); chunk 1 = +GSTR / +LSTR
    const char* gA = (const char*)(A  + (size_t)(m0 + rowL) * K + colL);
    const char* gB = (const char*)(Bw + (size_t)(n0 + rowL) * K + colL);
    const char* lA0 = (const char*)&As[0][rowL * LDA + colL];
    const char* lA1 = (const char*)&As[1][rowL * LDA + colL];
    const char* lB0 = (const char*)&Bs[0][rowL * LDA + colL];
    const char* lB1 = (const char*)&Bs[1][rowL * LDA + colL];

    f32x8 acc[4][2];
#pragma unroll
    for (int i = 0; i < 4; ++i)
#pragma unroll
        for (int j = 0; j < 2; ++j) acc[i][j] = (f32x8)(0.0f);

    auto issue = [&](unsigned kby, const char* lA, const char* lB) {
        async_b128(gA + kby,        lA);
        async_b128(gA + kby + GSTR, lA + LSTR);
        async_b128(gB + kby,        lB);
        async_b128(gB + kby + GSTR, lB + LSTR);
    };
    auto compute = [&](int b) {
        bf16x16 af[4], bfr[2];
#pragma unroll
        for (int i = 0; i < 4; ++i) af[i]  = load_frag(As[b], wm + i * 16 + lrow, kb, LDA);
#pragma unroll
        for (int j = 0; j < 2; ++j) bfr[j] = load_frag(Bs[b], wn + j * 16 + lrow, kb, LDA);
#pragma unroll
        for (int i = 0; i < 4; ++i)
#pragma unroll
            for (int j = 0; j < 2; ++j)
                acc[i][j] = __builtin_amdgcn_wmma_f32_16x16x32_bf16(
                    false, af[i], false, bfr[j], (short)0, acc[i][j], false, false);
    };

    issue(0, lA0, lB0);
    wait_async0();
    __syncthreads();

    unsigned kby = BK * 2;                 // byte offset of next tile
#pragma unroll 1
    for (int it = 0; it < K / BK / 2; ++it) {
        issue(kby, lA1, lB1); kby += BK * 2;   // last issue overshoots (padded ws)
        compute(0);
        wait_async0();
        __syncthreads();
        issue(kby, lA0, lB0); kby += BK * 2;
        compute(1);
        wait_async0();
        __syncthreads();
    }

    const int mhalf = (lane >> 4) * 8;
#pragma unroll
    for (int i = 0; i < 4; ++i)
#pragma unroll
        for (int j = 0; j < 2; ++j)
#pragma unroll
            for (int v = 0; v < 8; ++v) {
                int row = m0 + wm + i * 16 + mhalf + v;
                int col = n0 + wn + j * 16 + lrow;
                float val = acc[i][j][v];
                if constexpr (EPI == 0) {
                    outF[(size_t)row * ldOut + col] = val;
                } else if constexpr (EPI == 1) {
                    float x = val + bias[col];
                    float g = 0.5f * x * (1.0f + erff(x * 0.70710678118654752f));
                    outB[(size_t)row * ldOut + col] = f2bf(g);
                } else {
                    outF[((size_t)row * RH + sIdx) * DDIM + col] = val + bias[col];
                }
            }
}

// --------------------- fused GRU gate GEMM (64x384x32) ---------------------
// A = h_prev bf16 [1792,1024], Bw = gate-interleaved Wg [6144,1024].
// 8 waves = 2(M) x 4(N); wave tile 32x96 = 2 A-frags x 6 B-frags.
// Fragment fj == gate index; lane (j%16) owns all 6 gates of hidden unit j.

__global__ __launch_bounds__(256) void k_gru_gemm(
    const unsigned short* __restrict__ A,     // h_prev bf16
    const unsigned short* __restrict__ Wg,
    const float* __restrict__ cp,             // ctxproj [2048,3072]
    const float* __restrict__ b_ih,
    const float* __restrict__ b_hh,
    const float* __restrict__ hFin,
    float* __restrict__ hFout,
    unsigned short* __restrict__ hBout,
    int s)
{
    constexpr int BM = 64, BN = 384, BK = 32, LDA = BK + 8, K = GK;
    constexpr unsigned GSTR = 64u * K * 2u;
    constexpr unsigned LSTR = 64u * LDA * 2u;
    __shared__ __align__(16) unsigned short As[2][BM * LDA];
    __shared__ __align__(16) unsigned short Bs[2][BN * LDA];

    const int t     = threadIdx.x;
    const int m0    = blockIdx.y * BM;
    const int n0g   = blockIdx.x;             // group of 4 16-unit blocks
    const int lane  = t & 31;
    const int wave  = t >> 5;
    const int wm    = (wave >> 2) * 32;       // 2 waves along M
    const int waveN = wave & 3;               // 4 waves along N (96 cols each)
    const int wn    = waveN * 96;
    const int lrow  = lane & 15;
    const int kb    = (lane < 16) ? 0 : 8;

    const int rowA = t >> 2;                  // 0..63
    const int colL = (t & 3) * 8;

    const char* gA = (const char*)(A  + (size_t)(m0 + rowA) * K + colL);
    const char* gB = (const char*)(Wg + (size_t)(n0g * BN + rowA) * K + colL);
    const char* lA0 = (const char*)&As[0][rowA * LDA + colL];
    const char* lA1 = (const char*)&As[1][rowA * LDA + colL];
    const char* lB0 = (const char*)&Bs[0][rowA * LDA + colL];
    const char* lB1 = (const char*)&Bs[1][rowA * LDA + colL];

    f32x8 acc[2][6];
#pragma unroll
    for (int i = 0; i < 2; ++i)
#pragma unroll
        for (int j = 0; j < 6; ++j) acc[i][j] = (f32x8)(0.0f);

    auto issue = [&](unsigned kby, const char* lA, const char* lB) {
        async_b128(gA + kby, lA);
#pragma unroll
        for (int c = 0; c < 6; ++c)                  // 6 x 64-row chunks = 384 rows
            async_b128(gB + kby + c * GSTR, lB + c * LSTR);
    };
    auto compute = [&](int b) {
        bf16x16 af[2], bfr[6];
#pragma unroll
        for (int i = 0; i < 2; ++i) af[i]  = load_frag(As[b], wm + i * 16 + lrow, kb, LDA);
#pragma unroll
        for (int j = 0; j < 6; ++j) bfr[j] = load_frag(Bs[b], wn + j * 16 + lrow, kb, LDA);
#pragma unroll
        for (int i = 0; i < 2; ++i)
#pragma unroll
            for (int j = 0; j < 6; ++j)
                acc[i][j] = __builtin_amdgcn_wmma_f32_16x16x32_bf16(
                    false, af[i], false, bfr[j], (short)0, acc[i][j], false, false);
    };

    issue(0, lA0, lB0);
    wait_async0();
    __syncthreads();

    unsigned kby = BK * 2;
#pragma unroll 1
    for (int it = 0; it < K / BK / 2; ++it) {
        issue(kby, lA1, lB1); kby += BK * 2;
        compute(0);
        wait_async0();
        __syncthreads();
        issue(kby, lA0, lB0); kby += BK * 2;
        compute(1);
        wait_async0();
        __syncthreads();
    }

    // ---- fused GRU epilogue (all 6 gates of unit j live in this lane) ----
    const int j     = (n0g * 4 + waveN) * 16 + lrow;   // hidden unit index
    const int mhalf = (lane >> 4) * 8;
    const float bir = b_ih[j],        biz = b_ih[1024 + j], bin = b_ih[2048 + j];
    const float bhr = b_hh[j],        bhz = b_hh[1024 + j], bhn = b_hh[2048 + j];

#pragma unroll
    for (int i = 0; i < 2; ++i) {
#pragma unroll
        for (int v = 0; v < 8; ++v) {
            int m  = m0 + wm + i * 16 + mhalf + v;
            int b  = m / SC;
            int tt = m - b * SC;
            size_t cpr = (size_t)(b * TSEQ + tt + s + 1) * (3 * HDIM);

            float ir = acc[i][0][v] + cp[cpr + j]        + bir;
            float iz = acc[i][1][v] + cp[cpr + 1024 + j] + biz;
            float in = acc[i][2][v] + cp[cpr + 2048 + j] + bin;
            float hr = acc[i][3][v] + bhr;
            float hz = acc[i][4][v] + bhz;
            float hn = acc[i][5][v] + bhn;

            float r = sigmoidf_(ir + hr);
            float z = sigmoidf_(iz + hz);
            float n = tanhf(in + r * hn);
            float hprev = hFin[(size_t)m * HDIM + j];
            float hnew  = (1.0f - z) * n + z * hprev;
            hFout[(size_t)m * HDIM + j] = hnew;
            hBout[(size_t)m * HDIM + j] = f2bf(hnew);
        }
    }
}

// ------------------------------- launcher ----------------------------------

extern "C" void kernel_launch(void* const* d_in, const int* in_sizes, int n_in,
                              void* d_out, int out_size, void* d_ws, size_t ws_size,
                              hipStream_t stream) {
    const float* state = (const float*)d_in[0];
    const float* tctx  = (const float*)d_in[1];
    const float* W_ih  = (const float*)d_in[2];
    const float* W_hh  = (const float*)d_in[3];
    const float* b_ih  = (const float*)d_in[4];
    const float* b_hh  = (const float*)d_in[5];
    const float* W1    = (const float*)d_in[6];
    const float* b1    = (const float*)d_in[7];
    const float* W2    = (const float*)d_in[8];
    const float* b2    = (const float*)d_in[9];
    float* out = (float*)d_out;

    char* base = (char*)d_ws;
    size_t off = 0;
    // +1KB guard after every buffer: always-prefetch overshoots <=64B past end
    auto alloc = [&](size_t bytes) { char* p = base + off;
                                     off += (bytes + 1024 + 255) & ~(size_t)255; return p; };
    unsigned short* Wg   = (unsigned short*)alloc((size_t)6144 * 1024 * 2);
    unsigned short* Wc   = (unsigned short*)alloc((size_t)3072 * 1024 * 2);
    unsigned short* W1b  = (unsigned short*)alloc((size_t)1024 * 1024 * 2);
    unsigned short* W2b  = (unsigned short*)alloc((size_t)1024 * 1024 * 2);
    unsigned short* tcb  = (unsigned short*)alloc((size_t)BATCH * TSEQ * DDIM * 2);
    float*          cpj  = (float*)alloc((size_t)BATCH * TSEQ * 3072 * 4);
    float*          hFa  = (float*)alloc((size_t)MROWS * DDIM * 4);
    float*          hFb  = (float*)alloc((size_t)MROWS * DDIM * 4);
    unsigned short* hBa  = (unsigned short*)alloc((size_t)MROWS * DDIM * 2);
    unsigned short* hBb  = (unsigned short*)alloc((size_t)MROWS * DDIM * 2);
    unsigned short* midB = (unsigned short*)alloc((size_t)MROWS * DDIM * 2);
    (void)ws_size; (void)in_sizes; (void)n_in; (void)out_size;

    const int TPB = 256;
    // one-time conversions
    k_build_wg<<<(6 * HDIM * DDIM + TPB - 1) / TPB, TPB, 0, stream>>>(W_ih, W_hh, Wg);
    k_build_wc<<<(3 * HDIM * DDIM + TPB - 1) / TPB, TPB, 0, stream>>>(W_ih, Wc);
    k_f32_to_bf16<<<(DDIM * HDIM + TPB - 1) / TPB, TPB, 0, stream>>>(W1, W1b, DDIM * HDIM);
    k_f32_to_bf16<<<(DDIM * DDIM + TPB - 1) / TPB, TPB, 0, stream>>>(W2, W2b, DDIM * DDIM);
    k_f32_to_bf16<<<(BATCH * TSEQ * DDIM + TPB - 1) / TPB, TPB, 0, stream>>>(
        tctx, tcb, BATCH * TSEQ * DDIM);
    k_init_h<<<(MROWS * DDIM + TPB - 1) / TPB, TPB, 0, stream>>>(state, hFa, hBa);

    // context projection once: [2048,1024] x [1024,3072] -> cpj
    k_gemm<0><<<dim3(3072 / 128, (BATCH * TSEQ) / 128), 256, 0, stream>>>(
        tcb, Wc, cpj, nullptr, nullptr, BATCH * TSEQ, 3072, 3072, 0);

    float*          hF_cur = hFa; float*          hF_nxt = hFb;
    unsigned short* hB_cur = hBa; unsigned short* hB_nxt = hBb;

    for (int s = 0; s < RH; ++s) {
        // fused gate GEMM + GRU cell: h_prev -> h_new (ping-pong buffers)
        k_gru_gemm<<<dim3(6144 / 384, MROWS / 64), 256, 0, stream>>>(
            hB_cur, Wg, cpj, b_ih, b_hh, hF_cur, hF_nxt, hB_nxt, s);
        // mid = gelu(h_new @ W1^T + b1) -> bf16
        k_gemm<1><<<dim3(1024 / 128, MROWS / 128), 256, 0, stream>>>(
            hB_nxt, W1b, nullptr, midB, b1, MROWS, 1024, 1024, 0);
        // out = mid @ W2^T + b2 -> strided f32 store into (b,t,s,d)
        k_gemm<2><<<dim3(1024 / 128, MROWS / 128), 256, 0, stream>>>(
            midB, W2b, out, nullptr, b2, MROWS, 1024, 1024, s);

        float* tf = hF_cur; hF_cur = hF_nxt; hF_nxt = tf;
        unsigned short* tb = hB_cur; hB_cur = hB_nxt; hB_nxt = tb;
    }
}
